// CrossAttentionLayer_57294863729381
// MI455X (gfx1250) — matrix operand order repaired and checked by reading, not compiled, not run on hardware
//
#include <hip/hip_runtime.h>
#include <hip/hip_bf16.h>
#include <math.h>

// MI455X / gfx1250: wave32, WMMA 16x16x32 f16->f32, 320KB LDS per WGP.

typedef _Float16 h16v __attribute__((ext_vector_type(16)));
typedef _Float16 h8v  __attribute__((ext_vector_type(8)));
typedef float    f8v  __attribute__((ext_vector_type(8)));

#define DDIM 512
#define ATTN_SCALE 0.04419417382415922f  // 1/sqrt(512)

#if defined(__has_builtin)
#if __has_builtin(__builtin_amdgcn_global_load_async_to_lds_b128) && \
    __has_builtin(__builtin_amdgcn_s_wait_asynccnt)
#define CA_ASYNC_LDS 1
#endif
#endif

#if CA_ASYNC_LDS
// Exact param types per hipcc diagnostic: int4 in AS1 (global) / AS3 (LDS).
typedef int ca_i4 __attribute__((vector_size(4 * sizeof(int))));
typedef ca_i4 __attribute__((address_space(1)))* ca_gp4;
typedef ca_i4 __attribute__((address_space(3)))* ca_lp4;
#endif

__device__ __forceinline__ float warp_sum32(float v) {
  #pragma unroll
  for (int o = 16; o; o >>= 1) v += __shfl_xor(v, o, 32);
  return v;
}
__device__ __forceinline__ float warp_max32(float v) {
  #pragma unroll
  for (int o = 16; o; o >>= 1) v = fmaxf(v, __shfl_xor(v, o, 32));
  return v;
}

// f32 [B,N,D] -> f16 row-major copy + f16 transpose [B,D,N]
__global__ void ca_prep_kernel(const float* __restrict__ X,
                               _Float16* __restrict__ X16,
                               _Float16* __restrict__ XT16,
                               int N, long total) {
  long idx = (long)blockIdx.x * blockDim.x + threadIdx.x;
  if (idx >= total) return;
  int d = (int)(idx % DDIM);
  long nd = idx / DDIM;
  int n = (int)(nd % N);
  long b = nd / N;
  _Float16 h = (_Float16)X[idx];
  X16[idx] = h;
  XT16[(b * DDIM + d) * (long)N + n] = h;
}

// Fused masked-softmax cross attention for one 16-row query tile.
// Q16:[B,NQ,D] f16, K16:[B,NK,D] f16, VT16:[B,D,NK] f16 (V transposed), H16:[B,NQ,D] f16 out.
template<int NQ, int NK>
__global__ __launch_bounds__(128)
void ca_attn_kernel(const _Float16* __restrict__ Q16,
                    const _Float16* __restrict__ K16,
                    const _Float16* __restrict__ VT16,
                    _Float16* __restrict__ H16) {
  extern __shared__ char smem[];
  _Float16* Qt = (_Float16*)smem;                 // [16][DDIM]
  _Float16* S  = Qt + 16 * DDIM;                  // [16][NK] scores -> unnormalized probs
  float* rowinv = (float*)(S + 16 * NK);          // [16] per-row 1/sum

  const int tid  = threadIdx.x;
  const int lane = tid & 31;
  const int wave = tid >> 5;           // 4 waves
  const int ln   = lane & 15;
  const int hi   = lane >> 4;          // half-wave select
  const int q0   = blockIdx.x * 16;
  const long b   = blockIdx.y;

  // ---- phase 0: stage Q tile into LDS (f16, row-major) ----
  const _Float16* qsrc = Q16 + (b * NQ + q0) * (long)DDIM;
#if CA_ASYNC_LDS
  for (int i = tid; i < 16 * DDIM / 8; i += 128)
    __builtin_amdgcn_global_load_async_to_lds_b128(
        (ca_gp4)(qsrc + i * 8), (ca_lp4)(Qt + i * 8), 0, 0);
  __builtin_amdgcn_s_wait_asynccnt(0);
#else
  for (int i = tid; i < 16 * DDIM / 8; i += 128)
    ((h8v*)Qt)[i] = ((const h8v*)qsrc)[i];
#endif
  __syncthreads();

  // ---- phase 1: S = (Q K^T) * scale via WMMA, f16 rows into LDS ----
  const _Float16* kbase = K16 + b * (long)NK * DDIM;
  for (int nt = wave; nt < NK / 16; nt += 4) {
    const int n0 = nt * 16;
    if (nt + 4 < NK / 16)  // pull next tile's K rows toward the WGP (global_prefetch_b8)
      __builtin_prefetch(kbase + ((nt + 4) * 16 + ln) * (long)DDIM + hi * 256, 0, 1);
    f8v acc = {};
    const _Float16* krow = kbase + (n0 + ln) * (long)DDIM;
    for (int k0 = 0; k0 < DDIM; k0 += 32) {
      union { h16v v; h8v h[2]; } a, bm;
      const int aoff = k0 + hi * 8;
      a.h[0] = *(const h8v*)(Qt + ln * DDIM + aoff);
      a.h[1] = *(const h8v*)(Qt + ln * DDIM + aoff + 16);
      const int boff = k0 + hi * 8;
      bm.h[0] = *(const h8v*)(krow + boff);
      bm.h[1] = *(const h8v*)(krow + boff + 16);
      acc = __builtin_amdgcn_wmma_f32_16x16x32_f16(false, a.v, false, bm.v,
                                                   (short)0, acc, false, false);
    }
    #pragma unroll
    for (int r = 0; r < 8; ++r)
      S[(r + hi * 8) * NK + n0 + ln] = (_Float16)(acc[r] * ATTN_SCALE);
  }
  __syncthreads();

  // ---- phase 2: per-row mean/std(unbiased)/threshold + masked exp (vectorized, in LDS);
  //      normalization factor deferred to phase 3 via rowinv[] ----
  for (int r = wave * 4; r < wave * 4 + 4; ++r) {
    h8v* Srow = (h8v*)(S + r * NK);
    const int NV = NK / 8;
    float sum = 0.f, ss = 0.f;
    for (int k = lane; k < NV; k += 32) {
      h8v v8 = Srow[k];
      #pragma unroll
      for (int j = 0; j < 8; ++j) { float v = (float)v8[j]; sum += v; ss += v * v; }
    }
    sum = warp_sum32(sum); ss = warp_sum32(ss);
    const float mean = sum / (float)NK;
    float var = (ss - (float)NK * mean * mean) / (float)(NK - 1);
    var = fmaxf(var, 0.f);
    const float thr = mean + sqrtf(var) * 0.125f;

    float m = -1e30f;
    for (int k = lane; k < NV; k += 32) {
      h8v v8 = Srow[k];
      #pragma unroll
      for (int j = 0; j < 8; ++j) { float v = (float)v8[j]; if (v >= thr) m = fmaxf(m, v); }
    }
    m = warp_max32(m);

    float ps = 0.f;
    for (int k = lane; k < NV; k += 32) {
      h8v v8 = Srow[k];
      h8v p8;
      #pragma unroll
      for (int j = 0; j < 8; ++j) {
        float v = (float)v8[j];
        float p = (v >= thr) ? __expf(v - m) : 0.f;
        p8[j] = (_Float16)p;
        ps += p;
      }
      Srow[k] = p8;
    }
    ps = warp_sum32(ps);
    if (lane == 0) rowinv[r] = (ps > 0.f) ? 1.f / ps : 0.f;
  }
  __syncthreads();

  // ---- phase 3: H = diag(rowinv) * P * V via WMMA; B rows from V^T (contiguous) ----
  const _Float16* vtb = VT16 + b * (long)DDIM * NK;
  float rinv[8];
  #pragma unroll
  for (int r = 0; r < 8; ++r) rinv[r] = rowinv[r + hi * 8];

  f8v acc[8];
  #pragma unroll
  for (int t = 0; t < 8; ++t) acc[t] = (f8v){};
  for (int k0 = 0; k0 < NK; k0 += 32) {
    union { h16v v; h8v h[2]; } a;
    const int aoff = k0 + hi * 8;
    a.h[0] = *(const h8v*)(S + ln * NK + aoff);
    a.h[1] = *(const h8v*)(S + ln * NK + aoff + 16);
    #pragma unroll
    for (int t = 0; t < 8; ++t) {
      const int n0 = (wave * 8 + t) * 16;
      union { h16v v; h8v h[2]; } bm;
      const _Float16* vrow = vtb + (n0 + ln) * (long)NK + k0 + hi * 8;
      bm.h[0] = *(const h8v*)(vrow);
      bm.h[1] = *(const h8v*)(vrow + 16);
      acc[t] = __builtin_amdgcn_wmma_f32_16x16x32_f16(false, a.v, false, bm.v,
                                                      (short)0, acc[t], false, false);
    }
  }
  _Float16* hbase = H16 + (b * NQ + q0) * (long)DDIM;
  #pragma unroll
  for (int t = 0; t < 8; ++t) {
    const int n0 = (wave * 8 + t) * 16;
    #pragma unroll
    for (int r = 0; r < 8; ++r)
      hbase[(r + hi * 8) * (long)DDIM + n0 + ln] = (_Float16)(acc[t][r] * rinv[r]);
  }
}

// H_hybrid[B,4096,1024] f32 = Hp[B,4096,512] * Hc[B,1024,512]^T  (NT f16 WMMA GEMM)
__global__ __launch_bounds__(32)
void ca_hybrid_kernel(const _Float16* __restrict__ Hp,
                      const _Float16* __restrict__ Hc,
                      float* __restrict__ out) {
  const int lane = threadIdx.x & 31;
  const int ln = lane & 15, hi = lane >> 4;
  const int p0 = blockIdx.x * 16;
  const int c0 = blockIdx.y * 64;
  const long b = blockIdx.z;

  const _Float16* arow = Hp + (b * 4096 + p0 + ln) * (long)DDIM;
  f8v acc[4];
  #pragma unroll
  for (int t = 0; t < 4; ++t) acc[t] = (f8v){};
  for (int k0 = 0; k0 < DDIM; k0 += 32) {
    union { h16v v; h8v h[2]; } a;
    const int aoff = k0 + hi * 8;
    a.h[0] = *(const h8v*)(arow + aoff);
    a.h[1] = *(const h8v*)(arow + aoff + 16);
    #pragma unroll
    for (int t = 0; t < 4; ++t) {
      const _Float16* brow = Hc + (b * 1024 + c0 + t * 16 + ln) * (long)DDIM + k0 + hi * 8;
      union { h16v v; h8v h[2]; } bm;
      bm.h[0] = *(const h8v*)(brow);
      bm.h[1] = *(const h8v*)(brow + 16);
      acc[t] = __builtin_amdgcn_wmma_f32_16x16x32_f16(false, a.v, false, bm.v,
                                                      (short)0, acc[t], false, false);
    }
  }
  float* obase = out + (b * 4096 + p0) * (long)1024 + c0;
  #pragma unroll
  for (int t = 0; t < 4; ++t)
    #pragma unroll
    for (int r = 0; r < 8; ++r)
      obase[(r + hi * 8) * (long)1024 + t * 16 + ln] = acc[t][r];
}

extern "C" void kernel_launch(void* const* d_in, const int* in_sizes, int n_in,
                              void* d_out, int out_size, void* d_ws, size_t ws_size,
                              hipStream_t stream) {
  const float* Xp = (const float*)d_in[0];   // [4,4096,512] f32
  const float* Xc = (const float*)d_in[1];   // [4,1024,512] f32
  const int B = 4, NP = 4096, NC = 1024;

  char* ws = (char*)d_ws;
  size_t off = 0;
  auto carve = [&](size_t bytes) -> void* {
    void* p = ws + off;
    off += (bytes + 255) & ~(size_t)255;
    return p;
  };
  _Float16* Xp16  = (_Float16*)carve((size_t)B * NP * DDIM * 2);
  _Float16* XpT16 = (_Float16*)carve((size_t)B * NP * DDIM * 2);
  _Float16* Xc16  = (_Float16*)carve((size_t)B * NC * DDIM * 2);
  _Float16* XcT16 = (_Float16*)carve((size_t)B * NC * DDIM * 2);
  _Float16* Hp16  = (_Float16*)carve((size_t)B * NP * DDIM * 2);
  _Float16* Hc16  = (_Float16*)carve((size_t)B * NC * DDIM * 2);

  const long tp = (long)B * NP * DDIM;
  const long tc = (long)B * NC * DDIM;
  ca_prep_kernel<<<dim3((unsigned)((tp + 255) / 256)), 256, 0, stream>>>(Xp, Xp16, XpT16, NP, tp);
  ca_prep_kernel<<<dim3((unsigned)((tc + 255) / 256)), 256, 0, stream>>>(Xc, Xc16, XcT16, NC, tc);

  // patch -> channel:  NQ=4096, NK=1024; LDS = 16KB Q + 32KB S + 64B rowinv
  size_t sh_p2c = (size_t)16 * DDIM * 2 + (size_t)16 * 1024 * 2 + 16 * sizeof(float);
  ca_attn_kernel<4096, 1024><<<dim3(NP / 16, B), 128, sh_p2c, stream>>>(Xp16, Xc16, XcT16, Hp16);

  // channel -> patch:  NQ=1024, NK=4096; LDS = 16KB Q + 128KB S + 64B (fits 320KB WGP LDS)
  size_t sh_c2p = (size_t)16 * DDIM * 2 + (size_t)16 * 4096 * 2 + 16 * sizeof(float);
  ca_attn_kernel<1024, 4096><<<dim3(NC / 16, B), 128, sh_c2p, stream>>>(Xc16, Xp16, XpT16, Hc16);

  // hybrid fusion GEMM -> f32 output
  ca_hybrid_kernel<<<dim3(NP / 16, NC / 64, B), 32, 0, stream>>>(Hp16, Hc16, (float*)d_out);
}
